// LatentGRURNNAgent_89069031785155
// MI455X (gfx1250) — matrix operand order repaired and checked by reading, not compiled, not run on hardware
//
#include <hip/hip_runtime.h>

typedef unsigned short u16;
typedef unsigned int u32;
typedef __attribute__((ext_vector_type(16))) __bf16 v16bf;
typedef __attribute__((ext_vector_type(8)))  float  v8f;
typedef __attribute__((ext_vector_type(4)))  int    v4i;

#ifndef __has_builtin
#define __has_builtin(x) 0
#endif
#if __has_builtin(__builtin_amdgcn_global_load_async_to_lds_b128)
#define HAVE_ASYNC_LDS 1
#else
#define HAVE_ASYNC_LDS 0
#endif

union Frag16 {
    v16bf v;
    u16   u[16];
    uint4 q[2];
};
union BFC {
    __bf16 b;
    u16    u;
};

__device__ __forceinline__ u16 f2bf(float f) {
    BFC c;
    c.b = (__bf16)f;          // native cvt, RNE
    return c.u;
}
__device__ __forceinline__ float bf2f(u16 u) {
    BFC c;
    c.u = u;
    return (float)c.b;
}
__device__ __forceinline__ float sigf(float x) { return 1.0f / (1.0f + expf(-x)); }

__device__ __forceinline__ v8f wmma_bf16(const Frag16& a, const Frag16& b, v8f c) {
    return __builtin_amdgcn_wmma_f32_16x16x32_bf16(false, a.v, false, b.v, (short)0, c,
                                                   false, false);
}

// 16-byte global -> LDS copy: async-to-LDS on CDNA5, sync fallback otherwise
__device__ __forceinline__ void cp16(const u16* __restrict__ g, u16* l) {
#if HAVE_ASYNC_LDS
    __builtin_amdgcn_global_load_async_to_lds_b128(
        (__attribute__((address_space(1))) v4i*)g,
        (__attribute__((address_space(3))) v4i*)l, 0, 0);
#else
    *(uint4*)l = *(const uint4*)g;
#endif
}
__device__ __forceinline__ void wait_async_all() {
#if HAVE_ASYNC_LDS
#if __has_builtin(__builtin_amdgcn_s_wait_asynccnt)
    __builtin_amdgcn_s_wait_asynccnt(0);
#else
    asm volatile("s_wait_asynccnt 0" ::: "memory");
#endif
#endif
}

// ---------------------------------------------------------------------------
// fp32 -> bf16 conversion (plain, and transposing: [K][N] -> [N][K])
// ---------------------------------------------------------------------------
__global__ void conv_bf16_k(const float* __restrict__ s, u16* __restrict__ d, int n) {
    int i = blockIdx.x * 256 + threadIdx.x;
    if (i < n) d[i] = f2bf(s[i]);
}
__global__ void conv_bf16T_k(const float* __restrict__ s, u16* __restrict__ d, int K, int N) {
    int i = blockIdx.x * 256 + threadIdx.x;   // K*N total (multiple of 256)
    int k = i / N, n = i - k * N;
    d[(size_t)n * K + k] = f2bf(s[i]);
}

// ---------------------------------------------------------------------------
// Generic tiled bf16 WMMA GEMM: C[M,N] = A[M,K] @ B[K,N] (+bias)(+extra)(relu)
// A row-major [M][K] bf16; BT pre-transposed [N][K] bf16.
// block = 128 threads (4 waves), tile 64x64, K step 32
// ---------------------------------------------------------------------------
template <bool RELU, bool HAS_BIAS, bool HAS_EXTRA, bool WF32, bool WBF16>
__global__ __launch_bounds__(128) void gemm_bf16_k(
    const u16* __restrict__ A, const u16* __restrict__ BT,
    const float* __restrict__ bias, const float* __restrict__ extra,
    float* __restrict__ Cf, u16* __restrict__ Cb, int M, int N, int K) {
    const int bn0 = blockIdx.x * 64;
    const int bm0 = blockIdx.y * 64;
    const int t = threadIdx.x;
    const int wave = t >> 5, lane = t & 31;
    const int half = lane >> 4, r = lane & 15;

    __shared__ u16 As[64 * 40];   // [m][k], stride 40 (80B, 16B aligned)
    __shared__ u16 Bs[64 * 40];   // [n][k], stride 40

    v8f acc[4];
#pragma unroll
    for (int i = 0; i < 4; ++i)
#pragma unroll
        for (int j = 0; j < 8; ++j) acc[i][j] = 0.0f;

    const int row_l = wave * 16 + r;

    for (int kb = 0; kb < K; kb += 32) {
        __syncthreads();
        // stage A tile 64x32 and B tile 64x32 (both direct 16B copies)
#pragma unroll
        for (int i = 0; i < 2; ++i) {
            int q = t + i * 128;
            int m = q >> 2, koff = (q & 3) * 8;
            cp16(A + (size_t)(bm0 + m) * K + kb + koff, &As[m * 40 + koff]);
            cp16(BT + (size_t)(bn0 + m) * K + kb + koff, &Bs[m * 40 + koff]);
        }
        wait_async_all();
        __syncthreads();

        Frag16 af;
        af.q[0] = *(const uint4*)&As[row_l * 40 + half * 8];
        af.q[1] = *(const uint4*)&As[row_l * 40 + 16 + half * 8];
#pragma unroll
        for (int nt = 0; nt < 4; ++nt) {
            Frag16 bf_;
            int nl = nt * 16 + r;
            bf_.q[0] = *(const uint4*)&Bs[nl * 40 + half * 8];
            bf_.q[1] = *(const uint4*)&Bs[nl * 40 + 16 + half * 8];
            acc[nt] = wmma_bf16(af, bf_, acc[nt]);
        }
    }

#pragma unroll
    for (int nt = 0; nt < 4; ++nt) {
#pragma unroll
        for (int i = 0; i < 8; ++i) {
            int row = bm0 + wave * 16 + i + 8 * half;
            int col = bn0 + nt * 16 + r;
            float v = acc[nt][i];
            if (HAS_BIAS) v += bias[col];
            if (HAS_EXTRA) v += extra[(size_t)row * N + col];
            if (RELU) v = fmaxf(v, 0.0f);
            if (WF32) Cf[(size_t)row * N + col] = v;
            if (WBF16) Cb[(size_t)row * N + col] = f2bf(v);
        }
    }
}

// ---------------------------------------------------------------------------
// latent = mu + exp(logsig) * eps
// ---------------------------------------------------------------------------
__global__ void latent_k(const float* __restrict__ emb, const float* __restrict__ eps,
                         u16* __restrict__ latbf) {
    int i = blockIdx.x * 256 + threadIdx.x;   // B*64
    int b = i >> 6, j = i & 63;
    float mu = emb[(size_t)b * 128 + j];
    float ls = emb[(size_t)b * 128 + 64 + j];
    latbf[i] = f2bf(mu + expf(ls) * eps[i]);
}

// ---------------------------------------------------------------------------
// GRU gate combine: h = (1-z)*n + z*hprev
// ---------------------------------------------------------------------------
__global__ void gru_combine_k(const float* __restrict__ gx, const float* __restrict__ gh,
                              const float* __restrict__ hp, float* __restrict__ ho,
                              u16* __restrict__ hob, int W, int total) {
    int i = blockIdx.x * 256 + threadIdx.x;
    if (i >= total) return;
    int b = i / W, j = i - b * W;
    const float* gxr = gx + (size_t)b * 3 * W;
    const float* ghr = gh + (size_t)b * 3 * W;
    float r = sigf(gxr[j] + ghr[j]);
    float z = sigf(gxr[W + j] + ghr[W + j]);
    float n = tanhf(gxr[2 * W + j] + r * ghr[2 * W + j]);
    float h = (1.0f - z) * n + z * hp[i];
    ho[i] = h;
    if (hob) hob[i] = f2bf(h);
}

// ---------------------------------------------------------------------------
// loss: entropy + KL, per-block partials (deterministic), then softplus
// ---------------------------------------------------------------------------
__global__ void loss_reduce_k(const float* __restrict__ emb, const float* __restrict__ hist,
                              float* __restrict__ partials) {
    __shared__ float sd[256];
    int i = blockIdx.x * 256 + threadIdx.x;   // B*64
    int b = i >> 6, j = i & 63;
    float mu  = emb[(size_t)b * 128 + j];
    float ls  = emb[(size_t)b * 128 + 64 + j];
    float muh = hist[(size_t)b * 128 + j];
    float lsh = hist[(size_t)b * 128 + 64 + j];
    float ent = 1.4189385332046727f + ls;
    float d = mu - muh;
    float kl = (lsh - ls) + (expf(2.0f * ls) + d * d) * 0.5f * expf(-2.0f * lsh) - 0.5f;
    sd[threadIdx.x] = ent + kl;
    __syncthreads();
    for (int s = 128; s > 0; s >>= 1) {
        if (threadIdx.x < s) sd[threadIdx.x] += sd[threadIdx.x + s];
        __syncthreads();
    }
    if (threadIdx.x == 0) partials[blockIdx.x] = sd[0];
}

__global__ void loss_final_k(const float* __restrict__ partials, int nparts,
                             const int* __restrict__ t, float* __restrict__ out) {
    if (threadIdx.x == 0 && blockIdx.x == 0) {
        float s = 0.0f;
        for (int i = 0; i < nparts; ++i) s += partials[i];
        s /= 4096.0f;
        float loss = (t[0] != 0) ? ((s > 20.0f) ? s : log1pf(expf(s))) : 0.0f;
        out[0] = loss;
    }
}

// ---------------------------------------------------------------------------
// Fused q kernel: q[b,a] = sum_{hh,c} h[b,hh]*lat[b,c]*wnn_t[hh*64+a][c] + qpart
// block = 128 threads (4 waves), M-tile 32, N = 64, K = 256*256
// wave w: rows (w&1)*16..+15, n-tiles (w>>1)*32..+31
// A-fragment build is software-pipelined one K-chunk ahead so the
// v_pk_mul_bf16 batch never WARs against the in-flight WMMA pair.
// ---------------------------------------------------------------------------
__global__ __launch_bounds__(128) void q_fused_k(
    const u16* __restrict__ lat_bf,   // [4096][256]
    const u16* __restrict__ h_bf,     // [4096][256]
    const u16* __restrict__ wnn_t,    // [16384][256]
    const float* __restrict__ qpart,  // [4096][64]
    float* __restrict__ qout) {       // [4096][64]
    const int bm0 = blockIdx.x * 32;
    const int t = threadIdx.x;
    const int wave = t >> 5, lane = t & 31;
    const int half = lane >> 4, r = lane & 15;
    const int mtile = (wave & 1) << 4;
    const int mrow = mtile + r;            // local row 0..31
    const int nbase = (wave >> 1) << 5;    // 0 or 32

    __shared__ u16 Hs[32 * 264];   // [row][hh]
    __shared__ u16 Bs[64 * 264];   // [n][c]

    // lat fragments: register-resident for this lane's row, all 8 K-chunks
    Frag16 lf[8];
    const u16* lrow = lat_bf + (size_t)(bm0 + mrow) * 256;
#pragma unroll
    for (int cc = 0; cc < 8; ++cc) {
        lf[cc].q[0] = *(const uint4*)(lrow + cc * 32 + half * 8);
        lf[cc].q[1] = *(const uint4*)(lrow + cc * 32 + 16 + half * 8);
    }
    // stage h tile rows 0..15 (16 rows x 256)
#pragma unroll
    for (int i = 0; i < 4; ++i) {
        int q = t + i * 128;
        int row = q >> 5, off = (q & 31) * 8;
        cp16(h_bf + (size_t)(bm0 + row) * 256 + off, &Hs[row * 264 + off]);
    }

    v8f acc0, acc1;
#pragma unroll
    for (int j = 0; j < 8; ++j) { acc0[j] = 0.0f; acc1[j] = 0.0f; }

    for (int hh = 0; hh < 256; ++hh) {
        __syncthreads();
        const u16* wsrc = wnn_t + (size_t)hh * 16384;   // 64 rows x 256 = 32KB slab
#pragma unroll
        for (int i = 0; i < 16; ++i) {
            int q = t + i * 128;
            int n = q >> 5, off = (q & 31) * 8;
            cp16(wsrc + (size_t)n * 256 + off, &Bs[n * 264 + off]);
        }
        if (hh + 1 < 256)
            __builtin_prefetch(wsrc + 16384 + t * 128, 0, 0);
        wait_async_all();
        __syncthreads();

        // broadcast h[row, hh] as a packed-bf16 splat
        BFC hc;
        hc.u = Hs[mrow * 264 + hh];
        v16bf hs;
#pragma unroll
        for (int j = 0; j < 16; ++j) hs[j] = hc.b;

        // software-pipelined A-fragments (one chunk ahead)
        Frag16 af;
        af.v = lf[0].v * hs;
#pragma unroll
        for (int cc = 0; cc < 8; ++cc) {
            Frag16 afn;
            if (cc + 1 < 8) afn.v = lf[cc + 1].v * hs;   // next chunk, fresh regs
            Frag16 b0, b1;
            const int n0 = nbase + r, n1 = nbase + 16 + r;
            b0.q[0] = *(const uint4*)&Bs[n0 * 264 + cc * 32 + half * 8];
            b0.q[1] = *(const uint4*)&Bs[n0 * 264 + cc * 32 + 16 + half * 8];
            b1.q[0] = *(const uint4*)&Bs[n1 * 264 + cc * 32 + half * 8];
            b1.q[1] = *(const uint4*)&Bs[n1 * 264 + cc * 32 + 16 + half * 8];
            acc0 = wmma_bf16(af, b0, acc0);
            acc1 = wmma_bf16(af, b1, acc1);
            af = afn;
        }
    }

#pragma unroll
    for (int i = 0; i < 8; ++i) {
        int row = bm0 + mtile + i + 8 * half;
        int c0 = nbase + r, c1 = nbase + 16 + r;
        qout[(size_t)row * 64 + c0] = acc0[i] + qpart[(size_t)row * 64 + c0];
        qout[(size_t)row * 64 + c1] = acc1[i] + qpart[(size_t)row * 64 + c1];
    }
}

// ---------------------------------------------------------------------------
// host
// ---------------------------------------------------------------------------
extern "C" void kernel_launch(void* const* d_in, const int* in_sizes, int n_in,
                              void* d_out, int out_size, void* d_ws, size_t ws_size,
                              hipStream_t stream) {
    constexpr int B = 4096, D = 256, H = 256, A = 64;
    (void)in_sizes; (void)n_in; (void)out_size; (void)ws_size;

    const float* inputs      = (const float*)d_in[0];
    const float* hidden      = (const float*)d_in[1];
    const float* latent_last = (const float*)d_in[2];
    const float* latent_hist = (const float*)d_in[3];
    const float* eps         = (const float*)d_in[4];
    const float* embed_w     = (const float*)d_in[5];
    const float* embed_b     = (const float*)d_in[6];
    const float* lrnn_wih    = (const float*)d_in[7];
    const float* lrnn_bih    = (const float*)d_in[8];
    const float* lrnn_whh    = (const float*)d_in[9];
    const float* lrnn_bhh    = (const float*)d_in[10];
    const float* lfc1_w      = (const float*)d_in[11];
    const float* lfc1_b      = (const float*)d_in[12];
    const float* lfc2_w      = (const float*)d_in[13];
    const float* lfc2_b      = (const float*)d_in[14];
    const float* wnn_w       = (const float*)d_in[15];
    const float* wnn_b       = (const float*)d_in[16];
    const float* bnn_w       = (const float*)d_in[17];
    const float* bnn_b       = (const float*)d_in[18];
    const float* fc1_w       = (const float*)d_in[19];
    const float* fc1_b       = (const float*)d_in[20];
    const float* rnn_wih     = (const float*)d_in[21];
    const float* rnn_bih     = (const float*)d_in[22];
    const float* rnn_whh     = (const float*)d_in[23];
    const float* rnn_bhh     = (const float*)d_in[24];
    const int*   t_in        = (const int*)d_in[25];

    float* q_out = (float*)d_out;                                   // [B,A]
    float* h_out = (float*)d_out + (size_t)B * A;                   // [B,H]
    float* l_out = (float*)d_out + (size_t)B * A + (size_t)B * H;   // scalar

    char* base = (char*)d_ws;
    size_t off = 0;
    auto alloc = [&](size_t bytes) -> void* {
        void* p = base + off;
        off += (bytes + 255) & ~(size_t)255;
        return p;
    };

    // fp32 scratch
    float* emb   = (float*)alloc((size_t)B * 128 * 4);
    float* gxl   = (float*)alloc((size_t)B * 384 * 4);
    float* ghl   = (float*)alloc((size_t)B * 384 * 4);
    float* hist  = (float*)alloc((size_t)B * 128 * 4);
    float* gxh   = (float*)alloc((size_t)B * 768 * 4);
    float* ghh   = (float*)alloc((size_t)B * 768 * 4);
    float* fc2b  = (float*)alloc((size_t)B * 64 * 4);
    float* qpart = (float*)alloc((size_t)B * 64 * 4);
    float* parts = (float*)alloc(1024 * 4);
    // bf16 activations
    u16* xw_bf    = (u16*)alloc((size_t)B * D * 2);
    u16* hid_bf   = (u16*)alloc((size_t)B * H * 2);
    u16* llast_bf = (u16*)alloc((size_t)B * 128 * 2);
    u16* lhist_bf = (u16*)alloc((size_t)B * 128 * 2);
    u16* lat_in   = (u16*)alloc((size_t)B * 64 * 2);
    u16* lat1_bf  = (u16*)alloc((size_t)B * 256 * 2);
    u16* lat_bf   = (u16*)alloc((size_t)B * 256 * 2);
    u16* x_bf     = (u16*)alloc((size_t)B * 256 * 2);
    u16* h_bf     = (u16*)alloc((size_t)B * 256 * 2);
    // bf16 weights, pre-transposed to [N][K]
    u16* embw_t  = (u16*)alloc(128 * 256 * 2);
    u16* lwih_t  = (u16*)alloc(384 * 128 * 2);
    u16* lwhh_t  = (u16*)alloc(384 * 128 * 2);
    u16* lfc1_t  = (u16*)alloc(256 * 64 * 2);
    u16* lfc2_t  = (u16*)alloc(256 * 256 * 2);
    u16* bnnw_t  = (u16*)alloc(64 * 256 * 2);
    u16* fc1w_t  = (u16*)alloc(256 * 256 * 2);
    u16* rwih_t  = (u16*)alloc(768 * 256 * 2);
    u16* rwhh_t  = (u16*)alloc(768 * 256 * 2);
    u16* wnnb_t  = (u16*)alloc(64 * 256 * 2);
    u16* wnnT_bf = (u16*)alloc((size_t)16384 * 256 * 2);

    auto conv = [&](const float* s, u16* d, int n) {
        conv_bf16_k<<<(n + 255) / 256, 256, 0, stream>>>(s, d, n);
    };
    auto convT = [&](const float* s, u16* d, int K, int N) {
        conv_bf16T_k<<<(K * N) / 256, 256, 0, stream>>>(s, d, K, N);
    };
    conv(inputs, xw_bf, B * D);
    conv(hidden, hid_bf, B * H);
    conv(latent_last, llast_bf, B * 128);
    conv(latent_hist, lhist_bf, B * 128);
    convT(embed_w, embw_t, 256, 128);
    convT(lrnn_wih, lwih_t, 128, 384);
    convT(lrnn_whh, lwhh_t, 128, 384);
    convT(lfc1_w, lfc1_t, 64, 256);
    convT(lfc2_w, lfc2_t, 256, 256);
    convT(bnn_w, bnnw_t, 256, 64);
    convT(fc1_w, fc1w_t, 256, 256);
    convT(rnn_wih, rwih_t, 256, 768);
    convT(rnn_whh, rwhh_t, 256, 768);
    convT(wnn_b, wnnb_t, 256, 64);            // [256,64] -> [64][256]
    convT(wnn_w, wnnT_bf, 256, 16384);        // [256,16384] -> [16384][256]

    const dim3 blk(128);
    auto ggrid = [](int N) { return dim3(N / 64, B / 64); };

    gemm_bf16_k<false, true, false, true, false><<<ggrid(128), blk, 0, stream>>>(
        xw_bf, embw_t, embed_b, nullptr, emb, nullptr, B, 128, 256);
    latent_k<<<(B * 64) / 256, 256, 0, stream>>>(emb, eps, lat_in);
    gemm_bf16_k<false, true, false, true, false><<<ggrid(384), blk, 0, stream>>>(
        llast_bf, lwih_t, lrnn_bih, nullptr, gxl, nullptr, B, 384, 128);
    gemm_bf16_k<false, true, false, true, false><<<ggrid(384), blk, 0, stream>>>(
        lhist_bf, lwhh_t, lrnn_bhh, nullptr, ghl, nullptr, B, 384, 128);
    gru_combine_k<<<(B * 128) / 256, 256, 0, stream>>>(gxl, ghl, latent_hist, hist,
                                                       nullptr, 128, B * 128);
    loss_reduce_k<<<(B * 64) / 256, 256, 0, stream>>>(emb, hist, parts);
    gemm_bf16_k<true, true, false, false, true><<<ggrid(256), blk, 0, stream>>>(
        lat_in, lfc1_t, lfc1_b, nullptr, nullptr, lat1_bf, B, 256, 64);
    gemm_bf16_k<false, true, false, false, true><<<ggrid(256), blk, 0, stream>>>(
        lat1_bf, lfc2_t, lfc2_b, nullptr, nullptr, lat_bf, B, 256, 256);
    gemm_bf16_k<true, true, false, false, true><<<ggrid(256), blk, 0, stream>>>(
        xw_bf, fc1w_t, fc1_b, nullptr, nullptr, x_bf, B, 256, 256);
    gemm_bf16_k<false, true, false, true, false><<<ggrid(768), blk, 0, stream>>>(
        x_bf, rwih_t, rnn_bih, nullptr, gxh, nullptr, B, 768, 256);
    gemm_bf16_k<false, true, false, true, false><<<ggrid(768), blk, 0, stream>>>(
        hid_bf, rwhh_t, rnn_bhh, nullptr, ghh, nullptr, B, 768, 256);
    gru_combine_k<<<(B * 256) / 256, 256, 0, stream>>>(gxh, ghh, hidden, h_out, h_bf,
                                                       256, B * 256);
    gemm_bf16_k<false, true, false, true, false><<<ggrid(64), blk, 0, stream>>>(
        lat_bf, bnnw_t, bnn_b, nullptr, fc2b, nullptr, B, 64, 256);
    gemm_bf16_k<false, false, true, true, false><<<ggrid(64), blk, 0, stream>>>(
        h_bf, wnnb_t, nullptr, fc2b, qpart, nullptr, B, 64, 256);
    q_fused_k<<<B / 32, 128, 0, stream>>>(lat_bf, h_bf, wnnT_bf, qpart, q_out);
    loss_final_k<<<1, 64, 0, stream>>>(parts, 1024, t_in, l_out);
}
